// MultiScaleRetention_72533407695348
// MI455X (gfx1250) — matrix operand order repaired and checked by reading, not compiled, not run on hardware
//
#include <hip/hip_runtime.h>
#include <hip/hip_bf16.h>

typedef __attribute__((ext_vector_type(16))) _Float16 v16h;
typedef __attribute__((ext_vector_type(8)))  _Float16 v8h;
typedef __attribute__((ext_vector_type(8)))  float    v8f;

union AF { v16h v; v8h h[2]; };

#define HIDDEN 512
#define HD 64
#define NH 8
#define SEQ 2048
#define BATCH 4
#define ROWS (BATCH*SEQ)   // 8192
#define NALL 2048          // Q(512)|K(512)|V(512)|G(512) packed columns
#define NROT (SEQ*32)      // rotary table entries (s, j)

__device__ __forceinline__ v8f wmma16(v16h a, v16h b, v8f c) {
  return __builtin_amdgcn_wmma_f32_16x16x32_f16(false, a, false, b, (short)0, c, false, false);
}

// ------- Kernel 1: convert / pack (weights transposed N x K) + xPos rotary table -------
__global__ void pack_kernel(const float* __restrict__ X,  const float* __restrict__ WQ,
                            const float* __restrict__ WK, const float* __restrict__ WV,
                            const float* __restrict__ WG, const float* __restrict__ WO,
                            _Float16* __restrict__ Xh, _Float16* __restrict__ WallT,
                            _Float16* __restrict__ WoT, float4* __restrict__ Rot) {
  int idx = blockIdx.x * blockDim.x + threadIdx.x;
  const int NX = ROWS * HIDDEN;
  const int NWALL = NALL * HIDDEN;
  const int NWO = HIDDEN * HIDDEN;
  if (idx < NX) { Xh[idx] = (_Float16)X[idx]; return; }
  idx -= NX;
  if (idx < NWALL) {
    int n = idx >> 9, k = idx & 511;
    float v;
    if (n < 1536) {
      const float* W = (n < 512) ? WQ : (n < 1024) ? WK : WV;
      int h = (n >> 6) & 7, d = n & 63;
      v = W[(h * HIDDEN + k) * HD + d];          // W_* is [H][hidden][hd]
    } else {
      v = WG[k * HIDDEN + (n - 1536)];           // W_G is [hidden][512]
    }
    WallT[n * HIDDEN + k] = (_Float16)v;
    return;
  }
  idx -= NWALL;
  if (idx < NWO) {
    int n = idx >> 9, k = idx & 511;
    WoT[n * HIDDEN + k] = (_Float16)WO[k * HIDDEN + n];  // W_O is [512][512]
    return;
  }
  idx -= NWO;
  if (idx < NROT) {                              // xPos factors for pair j at position s
    int s = idx >> 5, j = idx & 31;
    float invf = __expf(-0.28782313662425572f * (float)j);   // ln(1e4)/32
    float ang = (float)s * invf;
    float sn, cs; __sincosf(ang, &sn, &cs);
    float sbase = ((float)(2 * j) + 25.6f) * (1.0f / 89.6f);
    float scl = __expf(__logf(sbase) * (float)s * (1.0f / 512.0f));
    float rscl = 1.0f / scl;
    Rot[idx] = make_float4(cs * scl, sn * scl, cs * rscl, sn * rscl);
  }
}

// ---------------- Kernel 2: fused QKV+gate GEMM with xPos / SiLU epilogue ----------------
// one wave computes a 32(M) x 64(N) tile of X @ Wall (B fragments reused across 2 M-subtiles)
__global__ void proj_kernel(const _Float16* __restrict__ Xh, const _Float16* __restrict__ WallT,
                            const float4* __restrict__ Rot,
                            _Float16* __restrict__ Qh, _Float16* __restrict__ Kh,
                            _Float16* __restrict__ VhT, _Float16* __restrict__ Gate) {
  const int lane = threadIdx.x & 31;
  const int half = lane >> 4;
  const int lm   = lane & 15;
  const int bi = blockIdx.x;
  const int m0 = (bi >> 5) << 5;                // 256 row tiles of 32
  const int n0 = (bi & 31) << 6;                // 32 col tiles of 64

  v8f acc[2][4] = {};
  const _Float16* arow0 = Xh + (m0 + lm) * HIDDEN;
  const _Float16* arow1 = Xh + (m0 + 16 + lm) * HIDDEN;
  for (int kk = 0; kk < HIDDEN; kk += 32) {
    AF a0, a1;
    a0.h[0] = *(const v8h*)(arow0 + kk + half * 8);
    a0.h[1] = *(const v8h*)(arow0 + kk + 16 + half * 8);
    a1.h[0] = *(const v8h*)(arow1 + kk + half * 8);
    a1.h[1] = *(const v8h*)(arow1 + kk + 16 + half * 8);
#pragma unroll
    for (int nt = 0; nt < 4; ++nt) {
      const _Float16* brow = WallT + (n0 + nt * 16 + lm) * HIDDEN + kk + half * 8;
      AF b; b.h[0] = *(const v8h*)brow; b.h[1] = *(const v8h*)(brow + 16);
      acc[0][nt] = wmma16(a0.v, b.v, acc[0][nt]);
      acc[1][nt] = wmma16(a1.v, b.v, acc[1][nt]);
    }
  }

  const int region = n0 >> 9;                   // whole block is one region
  const int hh = (n0 >> 6) & 7;
#pragma unroll
  for (int mi = 0; mi < 2; ++mi) {
#pragma unroll
    for (int nt = 0; nt < 4; ++nt) {
      int n = n0 + nt * 16 + lm;
      int d = n & 63;
#pragma unroll
      for (int cr = 0; cr < 8; ++cr) {
        int grow = m0 + mi * 16 + cr + 8 * half;
        int b = grow >> 11;
        int s = grow & (SEQ - 1);
        float x = acc[mi][nt][cr];
        if (region <= 1) {                      // Q or K: xPos rotary via table
          float px = __shfl_xor(x, 1, 32);      // rotary partner (adjacent column)
          float4 r = Rot[(s << 5) + (d >> 1)];
          float c  = (region == 0) ? r.x : r.z;
          float s_ = (region == 0) ? r.y : r.w;
          float outv = (d & 1) ? (x * c + px * s_) : (x * c - px * s_);
          int dst = ((b * NH + hh) * SEQ + s) * HD + d;
          ((region == 0) ? Qh : Kh)[dst] = (_Float16)outv;
        } else if (region == 2) {               // V: store transposed [bh][d][s]
          VhT[((b * NH + hh) * HD + d) * SEQ + s] = (_Float16)x;
        } else {                                // gate: SiLU
          float g = x * (1.0f / (1.0f + __expf(-x)));
          Gate[grow * HIDDEN + (n - 1536)] = (_Float16)g;
        }
      }
    }
  }
}

// ---------------- Kernel 3: chunk-masked retention (scores*decay then @V) ----------------
// one wave per (b, h, query-chunk, 16-row quarter); diagonal chunk first, then descend
// so the decay gamma^|i-j| becomes a single multiply-recurrence (w *= gamma^64).
__global__ void attn_kernel(const _Float16* __restrict__ Qh, const _Float16* __restrict__ Kh,
                            const _Float16* __restrict__ VhT, float* __restrict__ Yf) {
  __shared__ __align__(16) _Float16 sS[16 * 72];   // 16x64 scores, stride 72 (aligned + conflict-free)
  const int lane = threadIdx.x & 31;
  const int half = lane >> 4;
  const int lm   = lane & 15;
  const int bi = blockIdx.x;
  const int qq = bi & 3;
  const int qc = (bi >> 2) & 31;
  const int h  = (bi >> 7) & 7;
  const int b  = bi >> 10;
  const int bh = b * NH + h;
  const int s0 = qc * 64 + qq * 16;

  // gamma_h = 1 - exp(linspace(ln(1/32), ln(1/512), 8))[h]
  const float lgam = __logf(1.0f - __expf(-3.4657359028f - 0.3960857032f * (float)h));
  const float g64  = __expf(64.0f * lgam);         // gamma^64 (>= 5e-29 over full range: safe)

  v8f yacc[4] = {};
  float wreg[4][8];

  // Q fragments are invariant over the key loop: load once.
  const _Float16* qrow = Qh + (bh * SEQ + s0 + lm) * HD;
  AF qa[2];
#pragma unroll
  for (int kq = 0; kq < 2; ++kq) {
    qa[kq].h[0] = *(const v8h*)(qrow + kq * 32 + half * 8);
    qa[kq].h[1] = *(const v8h*)(qrow + kq * 32 + 16 + half * 8);
  }

  for (int step = 0; step <= qc; ++step) {
    const int kc = qc - step;                   // mask j < (qc+1)*64 handled structurally
    v8f sacc[4] = {};
#pragma unroll
    for (int kq = 0; kq < 2; ++kq) {
#pragma unroll
      for (int nt = 0; nt < 4; ++nt) {
        const _Float16* brow = Kh + (bh * SEQ + kc * 64 + nt * 16 + lm) * HD + kq * 32 + half * 8;
        AF bf; bf.h[0] = *(const v8h*)brow; bf.h[1] = *(const v8h*)(brow + 16);
        sacc[nt] = wmma16(qa[kq].v, bf.v, sacc[nt]);
      }
    }
    __syncthreads();
#pragma unroll
    for (int nt = 0; nt < 4; ++nt) {
#pragma unroll
      for (int cr = 0; cr < 8; ++cr) {
        int i = s0 + cr + 8 * half;
        int j = kc * 64 + nt * 16 + lm;
        float w;
        if (step == 0) {
          w = __expf(lgam * fabsf((float)(i - j)));     // diagonal chunk: j may exceed i
        } else if (step == 1) {
          w = __expf(lgam * (float)(i - j));            // seed off-diagonal weights
          wreg[nt][cr] = w;
        } else {
          w = wreg[nt][cr] * g64;                       // one chunk further: *= gamma^64
          wreg[nt][cr] = w;
        }
        sS[(cr + 8 * half) * 72 + nt * 16 + lm] = (_Float16)(sacc[nt][cr] * w);
      }
    }
    __syncthreads();
#pragma unroll
    for (int kk = 0; kk < 64; kk += 32) {
      AF a;
      const _Float16* srow = sS + lm * 72 + kk + half * 8;
      a.h[0] = *(const v8h*)(srow);
      a.h[1] = *(const v8h*)(srow + 16);
#pragma unroll
      for (int nt = 0; nt < 4; ++nt) {
        const _Float16* brow = VhT + (bh * HD + nt * 16 + lm) * SEQ + kc * 64 + kk + half * 8;
        AF bf; bf.h[0] = *(const v8h*)brow; bf.h[1] = *(const v8h*)(brow + 16);
        yacc[nt] = wmma16(a.v, bf.v, yacc[nt]);
      }
    }
  }
#pragma unroll
  for (int nt = 0; nt < 4; ++nt)
#pragma unroll
    for (int cr = 0; cr < 8; ++cr) {
      int row = s0 + cr + 8 * half;
      Yf[(bh * SEQ + row) * HD + nt * 16 + lm] = yacc[nt][cr];
    }
}

// ---------------- Kernel 4: per-head GroupNorm + gate ----------------
__global__ void norm_gate_kernel(const float* __restrict__ Yf, const _Float16* __restrict__ Gate,
                                 const float* __restrict__ gw, const float* __restrict__ gb,
                                 _Float16* __restrict__ Gn) {
  int row = blockIdx.x;                          // b*SEQ + s
  int b = row >> 11, s = row & (SEQ - 1);
  int h = threadIdx.x >> 5;
  int lane = threadIdx.x & 31;
  const float* yp = Yf + ((size_t)(b * NH + h) * SEQ + s) * HD;
  float y0 = yp[lane], y1 = yp[lane + 32];
  float sum = y0 + y1, sq = y0 * y0 + y1 * y1;
#pragma unroll
  for (int off = 16; off > 0; off >>= 1) {
    sum += __shfl_xor(sum, off, 32);
    sq  += __shfl_xor(sq,  off, 32);
  }
  float mean = sum * (1.0f / 64.0f);
  float var  = sq * (1.0f / 64.0f) - mean * mean;
  float rstd = rsqrtf(var + 1e-5f);
  int c0 = h * HD + lane, c1 = c0 + 32;
  float g0 = (float)Gate[row * HIDDEN + c0];
  float g1 = (float)Gate[row * HIDDEN + c1];
  float yn0 = (y0 - mean) * rstd * gw[c0] + gb[c0];
  float yn1 = (y1 - mean) * rstd * gw[c1] + gb[c1];
  Gn[row * HIDDEN + c0] = (_Float16)(g0 * yn0);
  Gn[row * HIDDEN + c1] = (_Float16)(g1 * yn1);
}

// ---------------- Kernel 5: output GEMM (Gn @ W_O) -> f32, 32x64 tile per wave ----------------
__global__ void out_kernel(const _Float16* __restrict__ Gn, const _Float16* __restrict__ WoT,
                           float* __restrict__ Out) {
  const int lane = threadIdx.x & 31;
  const int half = lane >> 4;
  const int lm   = lane & 15;
  const int bi = blockIdx.x;                    // 256 * 8 tiles
  const int m0 = (bi >> 3) << 5;
  const int n0 = (bi & 7) << 6;
  v8f acc[2][4] = {};
  const _Float16* arow0 = Gn + (m0 + lm) * HIDDEN;
  const _Float16* arow1 = Gn + (m0 + 16 + lm) * HIDDEN;
  for (int kk = 0; kk < HIDDEN; kk += 32) {
    AF a0, a1;
    a0.h[0] = *(const v8h*)(arow0 + kk + half * 8);
    a0.h[1] = *(const v8h*)(arow0 + kk + 16 + half * 8);
    a1.h[0] = *(const v8h*)(arow1 + kk + half * 8);
    a1.h[1] = *(const v8h*)(arow1 + kk + 16 + half * 8);
#pragma unroll
    for (int nt = 0; nt < 4; ++nt) {
      const _Float16* brow = WoT + (n0 + nt * 16 + lm) * HIDDEN + kk + half * 8;
      AF bf; bf.h[0] = *(const v8h*)brow; bf.h[1] = *(const v8h*)(brow + 16);
      acc[0][nt] = wmma16(a0.v, bf.v, acc[0][nt]);
      acc[1][nt] = wmma16(a1.v, bf.v, acc[1][nt]);
    }
  }
#pragma unroll
  for (int mi = 0; mi < 2; ++mi)
#pragma unroll
    for (int nt = 0; nt < 4; ++nt)
#pragma unroll
      for (int cr = 0; cr < 8; ++cr)
        Out[(m0 + mi * 16 + cr + 8 * half) * HIDDEN + n0 + nt * 16 + lm] = acc[mi][nt][cr];
}

extern "C" void kernel_launch(void* const* d_in, const int* in_sizes, int n_in,
                              void* d_out, int out_size, void* d_ws, size_t ws_size,
                              hipStream_t stream) {
  const float* X  = (const float*)d_in[0];
  const float* WQ = (const float*)d_in[1];
  const float* WK = (const float*)d_in[2];
  const float* WV = (const float*)d_in[3];
  const float* WG = (const float*)d_in[4];
  const float* WO = (const float*)d_in[5];
  const float* gw = (const float*)d_in[6];
  const float* gb = (const float*)d_in[7];
  float* Out = (float*)d_out;

  // workspace layout (~68 MB total)
  _Float16* Xh    = (_Float16*)d_ws;                         // 8192*512
  _Float16* WallT = Xh + (size_t)ROWS * HIDDEN;              // 2048*512
  _Float16* WoT   = WallT + (size_t)NALL * HIDDEN;           // 512*512
  _Float16* Qh    = WoT + (size_t)HIDDEN * HIDDEN;           // B*H*S*hd
  _Float16* Kh    = Qh + (size_t)BATCH * NH * SEQ * HD;
  _Float16* VhT   = Kh + (size_t)BATCH * NH * SEQ * HD;
  _Float16* Gate  = VhT + (size_t)BATCH * NH * SEQ * HD;     // 8192*512
  _Float16* Gn    = Gate + (size_t)ROWS * HIDDEN;            // 8192*512
  float*    Yf    = (float*)(Gn + (size_t)ROWS * HIDDEN);    // B*H*S*hd f32
  float4*   Rot   = (float4*)(Yf + (size_t)BATCH * NH * SEQ * HD);  // SEQ*32 float4 (1 MB)

  const int packN = ROWS * HIDDEN + NALL * HIDDEN + HIDDEN * HIDDEN + NROT;
  hipLaunchKernelGGL(pack_kernel, dim3((packN + 255) / 256), dim3(256), 0, stream,
                     X, WQ, WK, WV, WG, WO, Xh, WallT, WoT, Rot);
  hipLaunchKernelGGL(proj_kernel, dim3((ROWS / 32) * (NALL / 64)), dim3(32), 0, stream,
                     Xh, WallT, Rot, Qh, Kh, VhT, Gate);
  hipLaunchKernelGGL(attn_kernel, dim3(BATCH * NH * (SEQ / 64) * 4), dim3(32), 0, stream,
                     Qh, Kh, VhT, Yf);
  hipLaunchKernelGGL(norm_gate_kernel, dim3(ROWS), dim3(256), 0, stream,
                     Yf, Gate, gw, gb, Gn);
  hipLaunchKernelGGL(out_kernel, dim3((ROWS / 32) * (HIDDEN / 64)), dim3(32), 0, stream,
                     Gn, WoT, Out);
}
